// AttentionBlock_6038724019007
// MI455X (gfx1250) — matrix operand order repaired and checked by reading, not compiled, not run on hardware
//
#include <hip/hip_runtime.h>
#include <hip/hip_bf16.h>

// ---------------------------------------------------------------------------
// Problem constants (from reference): b=8, t=1024, n=256, h=8
// ---------------------------------------------------------------------------
static constexpr int Bz = 8;     // batch
static constexpr int Tz = 1024;  // sequence
static constexpr int Nz = 256;   // feature dim
static constexpr int Hz = 8;     // heads

typedef __attribute__((ext_vector_type(16))) _Float16 v16h;
typedef __attribute__((ext_vector_type(8)))  float    v8f;
typedef __attribute__((ext_vector_type(4)))  float    v4f;

// ---------------------------------------------------------------------------
// Fragment loaders for V_WMMA_F32_16X16X32_F16 (wave32 layouts per ISA 7.12.2)
//
// A (16x32, MxK): lane l -> row M = l%16; halves e=0..7  hold K = 8*(l/16)+e,
//                 halves e=8..15 hold K = 16 + 8*(l/16) + (e-8).
// B (32x16, KxN) stored as Bt[N][K] row-major: lane l -> col N = l%16,
//                 halves e hold K in the same order as A.
// ---------------------------------------------------------------------------
__device__ __forceinline__ v16h ld_a_frag(const _Float16* base, int m0, int ld,
                                          int k0, int lmod, int lhalf) {
  const _Float16* p = base + (size_t)(m0 + lmod) * ld + k0 + lhalf * 8;
  v16h a;
  ((v4f*)&a)[0] = *(const v4f*)(p);        // K = k0 + 8*lh + 0..7
  ((v4f*)&a)[1] = *(const v4f*)(p + 16);   // K = k0 + 16 + 8*lh + 0..7
  return a;
}

__device__ __forceinline__ v16h ld_b_frag(const _Float16* bt, int n0, int ld,
                                          int k0, int lmod, int lhalf) {
  const _Float16* p = bt + (size_t)(n0 + lmod) * ld + k0 + lhalf * 8;
  v16h b;
  ((v4f*)&b)[0] = *(const v4f*)(p);        // K = k0 + 8*lh + 0..7
  ((v4f*)&b)[1] = *(const v4f*)(p + 16);   // K = k0 + 16 + 8*lh + 0..7
  return b;
}

__device__ __forceinline__ v8f wmma16(v16h a, v16h b, v8f c) {
  return __builtin_amdgcn_wmma_f32_16x16x32_f16(
      /*neg_a=*/false, a, /*neg_b=*/false, b,
      /*c_mod=*/(short)0, c, /*reuse_a=*/false, /*reuse_b=*/false);
}

// ---------------------------------------------------------------------------
// Kernel 0: precision conversion + Q transpose.
//   s16[b,t,i]  = f16(s)
//   E16[h,i,j]  = f16(E)          (A operand of esT = E @ s^T)
//   Qt16[h,j,i] = f16(Q[h,i,j])   (Bt operand of qs = s @ Q)
// ---------------------------------------------------------------------------
__global__ void attn_convert(const float* __restrict__ s,
                             const float* __restrict__ Q,
                             const float* __restrict__ E,
                             _Float16* __restrict__ s16,
                             _Float16* __restrict__ Qt16,
                             _Float16* __restrict__ E16) {
  int i = blockIdx.x * blockDim.x + threadIdx.x;
  const int SN = Bz * Tz * Nz;          // 2,097,152
  const int QN = Hz * Nz * Nz;          // 524,288
  if (i < SN) {
    s16[i] = (_Float16)s[i];
  } else if (i < SN + QN) {
    int j = i - SN;
    int h   = j >> 16;                  // / (256*256)
    int rem = j & 65535;
    int r = rem >> 8, c = rem & 255;    // r = row i, c = col j of Q
    Qt16[h * 65536 + c * 256 + r] = (_Float16)Q[j];
    E16[j] = (_Float16)E[j];
  }
}

// ---------------------------------------------------------------------------
// Kernel 1: generic batched NT GEMM, K = 256 fixed, f16 in / f32 acc / f16 out
//   C[bh][m][n] = sum_k A[m][k] * Bt[n][k]
// Block: 256 threads = 8 waves (4 m-subtiles x 2 n-groups), tile 64m x 256n.
// ---------------------------------------------------------------------------
__global__ __launch_bounds__(256) void attn_gemm_nt(
    const _Float16* __restrict__ A, const _Float16* __restrict__ Bt,
    _Float16* __restrict__ C, int Ncols,
    int aStrB, int aStrH, int bStrB, int bStrH) {
  __shared__ alignas(16) _Float16 As[64 * 256];   // 32 KB A tile

  const int bh = blockIdx.z;
  const int bb = bh >> 3, hh = bh & 7;
  const _Float16* Ab  = A  + (size_t)bb * aStrB + (size_t)hh * aStrH;
  const _Float16* Btb = Bt + (size_t)bb * bStrB + (size_t)hh * bStrH;
  _Float16*       Cb  = C  + (size_t)bh * (Tz * Nz);

  const int m0 = blockIdx.y * 64;
  const int n0 = blockIdx.x * 256;
  const int tid = threadIdx.x;
  const int lane = tid & 31, wave = tid >> 5;
  const int lmod = lane & 15, lhalf = lane >> 4;
  const int msub = wave & 3, ngrp = wave >> 2;

  // Stage A tile [64][256] into LDS (8x 16B chunks per thread).
  for (int e = tid; e < 2048; e += 256) {
    int row = e >> 5;
    int c8  = (e & 31) * 8;
    *(v4f*)&As[row * 256 + c8] = *(const v4f*)&Ab[(size_t)(m0 + row) * 256 + c8];
  }
  __syncthreads();

  const v8f vzero = {};
  v8f acc[8];
#pragma unroll
  for (int g = 0; g < 8; ++g) acc[g] = vzero;

  const int nb = n0 + ngrp * 128;
#pragma unroll
  for (int f = 0; f < 8; ++f) {          // K = 256 = 8 * 32
    v16h a = ld_a_frag(As, msub * 16, 256, f * 32, lmod, lhalf);
#pragma unroll
    for (int g = 0; g < 8; ++g) {
      v16h b = ld_b_frag(Btb, nb + g * 16, 256, f * 32, lmod, lhalf);
      acc[g] = wmma16(a, b, acc[g]);
    }
  }

  // Store C as f16. D layout: vgpr r, lane l -> (M = r + 8*(l/16), N = l%16).
#pragma unroll
  for (int g = 0; g < 8; ++g) {
#pragma unroll
    for (int r = 0; r < 8; ++r) {
      int rr = m0 + msub * 16 + r + 8 * lhalf;
      int cc = nb + g * 16 + lmod;
      Cb[(size_t)rr * Ncols + cc] = (_Float16)acc[g][r];
    }
  }
}

// ---------------------------------------------------------------------------
// Kernel 2: fused masked Omega + r, one (b, h, 64-row t-tile) per block.
//   Omega[t][u] = sum_j qs[t][j] * s[u][j]      (causal: u <= t)
//   part[h][b][t][i] = sum_u Omega[t][u] * esT[i][u]
// 8 waves. Omega phase: wave w owns fragment (msub = w&3, usub = w>>2) of the
// 64x32 Omega slab -> all waves issue WMMAs, no idle phase, no branches.
// r phase: wave w owns rows msub*16, i-cols (w>>2)*128. Omega round-trips LDS
// to reshape D-layout -> A-layout.
// ---------------------------------------------------------------------------
__global__ __launch_bounds__(256) void attn_main(
    const _Float16* __restrict__ s16, const _Float16* __restrict__ QS,
    const _Float16* __restrict__ ESt, float* __restrict__ part) {
  __shared__ alignas(16) _Float16 qs_t[64 * 256];  // 32 KB
  __shared__ alignas(16) _Float16 s_u[32 * 256];   // 16 KB
  __shared__ alignas(16) _Float16 om[64 * 32];     //  4 KB

  const int tb = blockIdx.x * 64;
  const int bb = blockIdx.y;
  const int hh = blockIdx.z;
  const int bh = bb * 8 + hh;

  const int tid = threadIdx.x;
  const int lane = tid & 31, wave = tid >> 5;
  const int lmod = lane & 15, lhalf = lane >> 4;
  const int msub = wave & 3, igrp = wave >> 2;  // igrp doubles as Omega usub

  const _Float16* qsb  = QS  + (size_t)bh * (Tz * Nz);
  const _Float16* estb = ESt + (size_t)bh * (Tz * Nz);   // [i=256][u=1024]

  // Stage qs tile rows [tb, tb+64).
  for (int e = tid; e < 2048; e += 256) {
    int row = e >> 5;
    int c8  = (e & 31) * 8;
    *(v4f*)&qs_t[row * 256 + c8] =
        *(const v4f*)&qsb[(size_t)(tb + row) * 256 + c8];
  }
  __syncthreads();

  const v8f vzero = {};
  v8f racc[8];
#pragma unroll
  for (int g = 0; g < 8; ++g) racc[g] = vzero;

  const int uend = tb + 64;  // causal: only u-slabs with u0 <= t_max
  for (int u0 = 0; u0 < uend; u0 += 32) {
    // Stage s rows [u0, u0+32) (4x 16B chunks per thread).
    for (int e = tid; e < 1024; e += 256) {
      int row = e >> 5;
      int c8  = (e & 31) * 8;
      *(v4f*)&s_u[row * 256 + c8] =
          *(const v4f*)&s16[((size_t)bb * Tz + u0 + row) * 256 + c8];
    }
    if (u0 + 32 < uend)  // hint next s slab toward the caches
      __builtin_prefetch(&s16[((size_t)bb * Tz + u0 + 32 + (tid >> 3)) * 256], 0, 1);
    __syncthreads();

    // Omega fragment (msub, usub=igrp): all 8 waves participate.
    {
      v8f o = vzero;
#pragma unroll
      for (int f = 0; f < 8; ++f) {
        v16h a = ld_a_frag(qs_t, msub * 16, 256, f * 32, lmod, lhalf);
        v16h b = ld_b_frag(s_u, igrp * 16, 256, f * 32, lmod, lhalf);
        o = wmma16(a, b, o);
      }
      // Causal mask + f32->f16, write 16x16 Omega fragment to LDS.
      const int trow = tb + msub * 16 + 8 * lhalf;
      const int ucol = u0 + igrp * 16 + lmod;
#pragma unroll
      for (int r = 0; r < 8; ++r) {
        float w = (ucol <= trow + r) ? o[r] : 0.f;
        om[(msub * 16 + r + 8 * lhalf) * 32 + igrp * 16 + lmod] = (_Float16)w;
      }
    }
    __syncthreads();

    // r += Omega(16t x 32u) @ esT^T : B frags stream from L2-resident ESt.
    v16h a = ld_a_frag(om, msub * 16, 32, 0, lmod, lhalf);
#pragma unroll
    for (int g = 0; g < 8; ++g) {
      v16h b = ld_b_frag(estb, igrp * 128 + g * 16, Tz, u0, lmod, lhalf);
      racc[g] = wmma16(a, b, racc[g]);
    }
    __syncthreads();  // protect s_u / om before next slab
  }

  // Per-head partial, f32.
  float* pb = part + (size_t)hh * (Bz * Tz * Nz) + (size_t)bb * (Tz * Nz);
#pragma unroll
  for (int g = 0; g < 8; ++g) {
#pragma unroll
    for (int r = 0; r < 8; ++r) {
      int t = tb + msub * 16 + r + 8 * lhalf;
      int i = igrp * 128 + g * 16 + lmod;
      pb[(size_t)t * Nz + i] = racc[g][r];
    }
  }
}

// ---------------------------------------------------------------------------
// Kernel 3: reduce the 8 per-head partials into r (deterministic, no atomics).
// ---------------------------------------------------------------------------
__global__ void attn_reduce(const float* __restrict__ part,
                            float* __restrict__ out) {
  int i = blockIdx.x * blockDim.x + threadIdx.x;
  const int NE = Bz * Tz * Nz;
  if (i < NE) {
    float acc = 0.f;
#pragma unroll
    for (int h = 0; h < Hz; ++h) acc += part[(size_t)h * NE + i];
    out[i] = acc;
  }
}

// ---------------------------------------------------------------------------
// Host-side launcher. Workspace layout (134 MiB total):
//   [0,4MiB)    s16   f16[b][t][n]
//   [4,5MiB)    Qt16  f16[h][j][i]
//   [5,6MiB)    E16   f16[h][i][j]
//   [6,38MiB)   QS    f16[b,h][t][j]
//   [38,70MiB)  ESt   f16[b,h][i][u]
//   [70,134MiB) part  f32[h][b][t][i]
// ---------------------------------------------------------------------------
extern "C" void kernel_launch(void* const* d_in, const int* in_sizes, int n_in,
                              void* d_out, int out_size, void* d_ws, size_t ws_size,
                              hipStream_t stream) {
  (void)in_sizes; (void)n_in; (void)out_size; (void)ws_size;
  const float* s = (const float*)d_in[0];
  const float* Q = (const float*)d_in[1];
  const float* E = (const float*)d_in[2];
  float* out = (float*)d_out;

  char* ws = (char*)d_ws;
  const size_t MiB = 1ull << 20;
  _Float16* s16  = (_Float16*)(ws + 0);
  _Float16* Qt16 = (_Float16*)(ws + 4 * MiB);
  _Float16* E16  = (_Float16*)(ws + 5 * MiB);
  _Float16* QS   = (_Float16*)(ws + 6 * MiB);
  _Float16* ESt  = (_Float16*)(ws + 38 * MiB);
  float*    part = (float*)   (ws + 70 * MiB);

  // 0) convert / transpose
  {
    int total = Bz * Tz * Nz + Hz * Nz * Nz;       // 2,621,440
    attn_convert<<<(total + 255) / 256, 256, 0, stream>>>(s, Q, E, s16, Qt16, E16);
  }
  // 1a) QS[bh] = s16[b] @ Qt16[h]^T   (M=1024, N=256, K=256)
  attn_gemm_nt<<<dim3(1, Tz / 64, Bz * Hz), 256, 0, stream>>>(
      s16, Qt16, QS, /*Ncols=*/Nz,
      /*aStrB=*/Tz * Nz, /*aStrH=*/0, /*bStrB=*/0, /*bStrH=*/Nz * Nz);
  // 1b) ESt[bh] = E16[h] @ s16[b]^T   (M=256, N=1024, K=256)
  attn_gemm_nt<<<dim3(Tz / 256, Nz / 64, Bz * Hz), 256, 0, stream>>>(
      E16, s16, ESt, /*Ncols=*/Tz,
      /*aStrB=*/0, /*aStrH=*/Nz * Nz, /*bStrB=*/Tz * Nz, /*bStrH=*/0);
  // 2) fused masked Omega + r partials (one head per block for occupancy)
  attn_main<<<dim3(Tz / 64, Bz, Hz), 256, 0, stream>>>(s16, QS, ESt, part);
  // 3) head reduction
  {
    int total = Bz * Tz * Nz;
    attn_reduce<<<(total + 255) / 256, 256, 0, stream>>>(part, out);
  }
}
